// KGPathReasoner_19645180411921
// MI455X (gfx1250) — compile-verified, matrix operand order checked by reading, not run on hardware
//
#include <hip/hip_runtime.h>
#include <hip/hip_bf16.h>

// ---------------------------------------------------------------------------
// KG path reasoner forward for gfx1250 (MI455X, wave32, WMMA).
// All GEMMs use v_wmma_f32_16x16x32_f16 with LDS-staged 128x128x32 tiles.
// ---------------------------------------------------------------------------

typedef __attribute__((ext_vector_type(16))) _Float16 v16h;
typedef __attribute__((ext_vector_type(8)))  _Float16 v8h;
typedef __attribute__((ext_vector_type(8)))  float    v8f;

#define P_PAIRS 2048
#define K_PATHS 10
#define L_STEPS 3
#define E_DIM   256
#define H_DIM   512
#define NPK     (P_PAIRS * K_PATHS)   // 20480

#define TILE_M 128
#define TILE_N 128
#define TILE_K 32
#define LDS_A_STRIDE 40   // 32 halfs + 8 pad
#define LDS_B_STRIDE 40

// ---------------------------------------------------------------------------
// Generic f16 WMMA GEMM: C[M,N] = gather(A)[M,K] * B[K,N] + bias
//   A : row-major f16, lda in elements. Optional per-row gather indices.
//   B : row-major f16 [K,N] (already transposed on host-prep side).
//   out: either f16 (outH) or f32 (outF), row stride ldOut, column offset
//        colOff (for packing into wider concatenated buffers).
// Block: 256 threads = 8 waves; wave (wM,wN) in 4x2 grid; each wave does a
// 32x64 sub-tile = 2x4 accumulators of 16x16.
// ---------------------------------------------------------------------------
__global__ __launch_bounds__(256)
void wmma_gemm_f16(const _Float16* __restrict__ A, int lda,
                   const int* __restrict__ gather, int gStride, int gOff,
                   const _Float16* __restrict__ B, int ldb,
                   const float* __restrict__ bias0,
                   const float* __restrict__ bias1,
                   _Float16* __restrict__ outH, float* __restrict__ outF,
                   int ldOut, int colOff, int Kdim)
{
    __shared__ _Float16 As[TILE_M * LDS_A_STRIDE];
    __shared__ _Float16 Bs[TILE_N * LDS_B_STRIDE];   // transposed: Bs[n][k]

    const int tid   = threadIdx.x;
    const int lane  = tid & 31;
    const int wave  = tid >> 5;
    const int waveM = wave & 3;    // 0..3 -> 32-row slots
    const int waveN = wave >> 2;   // 0..1 -> 64-col slots
    const int m0 = blockIdx.y * TILE_M;
    const int n0 = blockIdx.x * TILE_N;

    const int rC      = lane & 15;       // row/col within 16x16 fragment
    const int halfSel = lane >> 4;       // 0: lanes 0-15, 1: lanes 16-31
    const int kSel    = halfSel * 8;     // K phase per ISA 16-bit A/B layout

    v8f acc[2][4] = {};

    for (int kc = 0; kc < Kdim; kc += TILE_K) {
        __syncthreads();
        // ---- A tile -> LDS (two passes of 64 rows, 16B vector loads) ----
        #pragma unroll
        for (int pass = 0; pass < 2; ++pass) {
            int r = pass * 64 + (tid >> 2);
            int c = (tid & 3) * 8;
            int gm = m0 + r;
            int arow = gather ? gather[(size_t)gm * gStride + gOff] : gm;
            const v8h* src = (const v8h*)(A + (size_t)arow * lda + kc + c);
            *(v8h*)(&As[r * LDS_A_STRIDE + c]) = *src;
        }
        // ---- B tile -> LDS transposed (Bs[n][k]) ----
        #pragma unroll
        for (int pass = 0; pass < 2; ++pass) {
            int k = pass * 16 + (tid >> 4);
            int c = (tid & 15) * 8;
            const v8h v = *(const v8h*)(B + (size_t)(kc + k) * ldb + n0 + c);
            #pragma unroll
            for (int e = 0; e < 8; ++e)
                Bs[(c + e) * LDS_B_STRIDE + k] = v[e];
        }
        if (kc + TILE_K < Kdim)   // hint next B tile into cache
            __builtin_prefetch(B + (size_t)(kc + TILE_K) * ldb + n0 +
                               (size_t)(tid & 127) * 8, 0, 1);
        __syncthreads();

        // ---- fragments (ISA 16-bit 16x32 layout: two 16B LDS reads each) --
        v16h aF[2], bF[4];
        #pragma unroll
        for (int mt = 0; mt < 2; ++mt) {
            int r = waveM * 32 + mt * 16 + rC;
            v8h lo = *(const v8h*)(&As[r * LDS_A_STRIDE + kSel]);
            v8h hi = *(const v8h*)(&As[r * LDS_A_STRIDE + kSel + 16]);
            #pragma unroll
            for (int e = 0; e < 8; ++e) { aF[mt][e] = lo[e]; aF[mt][e + 8] = hi[e]; }
        }
        #pragma unroll
        for (int nt = 0; nt < 4; ++nt) {
            int n = waveN * 64 + nt * 16 + rC;
            v8h lo = *(const v8h*)(&Bs[n * LDS_B_STRIDE + kSel]);
            v8h hi = *(const v8h*)(&Bs[n * LDS_B_STRIDE + kSel + 16]);
            #pragma unroll
            for (int e = 0; e < 8; ++e) { bF[nt][e] = lo[e]; bF[nt][e + 8] = hi[e]; }
        }
        #pragma unroll
        for (int mt = 0; mt < 2; ++mt)
            #pragma unroll
            for (int nt = 0; nt < 4; ++nt)
                acc[mt][nt] = __builtin_amdgcn_wmma_f32_16x16x32_f16(
                    false, aF[mt], false, bF[nt],
                    (short)0, acc[mt][nt], false, false);
    }

    // ---- epilogue: bias + store (C layout: col = lane&15, row = r+8*half) -
    #pragma unroll
    for (int nt = 0; nt < 4; ++nt) {
        int n = n0 + waveN * 64 + nt * 16 + rC;
        float bv = 0.0f;
        if (bias0) bv = bias0[n];
        if (bias1) bv += bias1[n];
        #pragma unroll
        for (int mt = 0; mt < 2; ++mt) {
            int mBase = m0 + waveM * 32 + mt * 16 + halfSel * 8;
            #pragma unroll
            for (int r = 0; r < 8; ++r) {
                float v = acc[mt][nt][r] + bv;
                size_t off = (size_t)(mBase + r) * ldOut + colOff + n;
                if (outH) outH[off] = (_Float16)v;
                else      outF[off] = v;
            }
        }
    }
}

// ---------------------------------------------------------------------------
// Prep kernels
// ---------------------------------------------------------------------------
__global__ void cvt_f32_f16(const float* __restrict__ in,
                            _Float16* __restrict__ out, int n) {
    int i = blockIdx.x * blockDim.x + threadIdx.x;
    if (i < n) out[i] = (_Float16)in[i];
}

// in [R,C] f32 row-major -> out [C,R] f16 (weight transpose for B operand)
__global__ void tcvt_f32_f16(const float* __restrict__ in,
                             _Float16* __restrict__ out, int R, int C) {
    int i = blockIdx.x * blockDim.x + threadIdx.x;
    if (i < R * C) {
        int r = i / C, c = i % C;
        out[(size_t)c * R + r] = (_Float16)in[i];
    }
}

// Stack [w_ih ; w_hh] transposed -> B[1536][2048]
__global__ void build_lstm_B(const float* __restrict__ w_ih,
                             const float* __restrict__ w_hh,
                             _Float16* __restrict__ out) {
    int i = blockIdx.x * blockDim.x + threadIdx.x;
    if (i < 1536 * 2048) {
        int k = i / 2048, nn = i % 2048;
        float v = (k < 1024) ? w_ih[(size_t)nn * 1024 + k]
                             : w_hh[(size_t)nn * 512 + (k - 1024)];
        out[i] = (_Float16)v;
    }
}

__global__ void vadd_f32(const float* a, const float* b, float* out, int n) {
    int i = blockIdx.x * blockDim.x + threadIdx.x;
    if (i < n) out[i] = a[i] + b[i];
}

// zero h-region of xh, cell state c, and path_emb
__global__ void init_state(_Float16* __restrict__ xh, float* __restrict__ c,
                           _Float16* __restrict__ pemb) {
    int i = blockIdx.x * blockDim.x + threadIdx.x;
    if (i < NPK * H_DIM) {
        int n = i >> 9, j = i & 511;
        xh[(size_t)n * 1536 + 1024 + j] = (_Float16)0.0f;
        c[i] = 0.0f;
        pemb[i] = (_Float16)0.0f;
    }
}

// ---------------------------------------------------------------------------
// LSTM cell (torch gate order i,f,g,o). Writes h into xh[:,1024:1536] (next
// step's A operand) and captures h_last into path_emb per path_lens.
// ---------------------------------------------------------------------------
__global__ void lstm_cell(const _Float16* __restrict__ g,
                          float* __restrict__ c,
                          _Float16* __restrict__ xh,
                          _Float16* __restrict__ pemb,
                          const int* __restrict__ path_lens, int t) {
    int i = blockIdx.x * blockDim.x + threadIdx.x;
    if (i >= NPK * H_DIM) return;
    int n = i >> 9, j = i & 511;
    const _Float16* gr = g + (size_t)n * 2048;
    float gi = (float)gr[j];
    float gf = (float)gr[512 + j];
    float gg = (float)gr[1024 + j];
    float go = (float)gr[1536 + j];
    float si = 1.0f / (1.0f + expf(-gi));
    float sf = 1.0f / (1.0f + expf(-gf));
    float so = 1.0f / (1.0f + expf(-go));
    float cn = sf * c[i] + si * tanhf(gg);
    c[i] = cn;
    float h = so * tanhf(cn);
    xh[(size_t)n * 1536 + 1024 + j] = (_Float16)h;
    int len = path_lens[n];
    int last = len - 1;
    if (last < 0) last = 0;
    if (last > L_STEPS - 1) last = L_STEPS - 1;
    if (len > 0 && last == t) pemb[(size_t)n * H_DIM + j] = (_Float16)h;
}

// ---------------------------------------------------------------------------
// Per-(pair, head) attention over K=10 paths, Dh=128. qkv rows are
// [q(512) | k(512) | v(512)], head h occupies cols h*128..h*128+127.
// ---------------------------------------------------------------------------
__global__ __launch_bounds__(128)
void attn_kernel(const _Float16* __restrict__ qkv, _Float16* __restrict__ ctx) {
    int p = blockIdx.x;      // pair
    int h = blockIdx.y;      // head
    int d = threadIdx.x;     // 0..127
    __shared__ float qs[K_PATHS][128], ks[K_PATHS][128], vs[K_PATHS][128];
    __shared__ float sc[K_PATHS][12];

    for (int i = 0; i < K_PATHS; ++i) {
        size_t row = (size_t)(p * K_PATHS + i) * 1536 + h * 128 + d;
        qs[i][d] = (float)qkv[row];
        ks[i][d] = (float)qkv[row + 512];
        vs[i][d] = (float)qkv[row + 1024];
    }
    __syncthreads();
    if (threadIdx.x < K_PATHS * K_PATHS) {
        int i = threadIdx.x / K_PATHS, j = threadIdx.x % K_PATHS;
        float s = 0.0f;
        for (int e = 0; e < 128; ++e) s += qs[i][e] * ks[j][e];
        sc[i][j] = s * 0.08838834764831845f;   // 1/sqrt(128)
    }
    __syncthreads();
    if (threadIdx.x < K_PATHS) {
        int i = threadIdx.x;
        float mx = sc[i][0];
        for (int j = 1; j < K_PATHS; ++j) mx = fmaxf(mx, sc[i][j]);
        float sum = 0.0f;
        for (int j = 0; j < K_PATHS; ++j) {
            float e = expf(sc[i][j] - mx);
            sc[i][j] = e; sum += e;
        }
        float inv = 1.0f / sum;
        for (int j = 0; j < K_PATHS; ++j) sc[i][j] *= inv;
    }
    __syncthreads();
    for (int i = 0; i < K_PATHS; ++i) {
        float a = 0.0f;
        for (int j = 0; j < K_PATHS; ++j) a += sc[i][j] * vs[j][d];
        ctx[(size_t)(p * K_PATHS + i) * H_DIM + h * 128 + d] = (_Float16)a;
    }
}

// mean over K paths
__global__ void mean_k(const _Float16* __restrict__ ao,
                       _Float16* __restrict__ agg) {
    int i = blockIdx.x * blockDim.x + threadIdx.x;
    if (i >= P_PAIRS * H_DIM) return;
    int p = i >> 9, j = i & 511;
    float s = 0.0f;
    for (int k = 0; k < K_PATHS; ++k)
        s += (float)ao[(size_t)(p * K_PATHS + k) * H_DIM + j];
    agg[i] = (_Float16)(s * 0.1f);
}

// ---------------------------------------------------------------------------
extern "C" void kernel_launch(void* const* d_in, const int* in_sizes, int n_in,
                              void* d_out, int out_size, void* d_ws, size_t ws_size,
                              hipStream_t stream) {
    (void)in_sizes; (void)n_in; (void)out_size; (void)ws_size;
    const int*   rel_idx     = (const int*)d_in[0];
    const int*   ent_idx     = (const int*)d_in[1];
    const int*   path_lens   = (const int*)d_in[2];
    const float* rel_table   = (const float*)d_in[3];
    const float* ent_table   = (const float*)d_in[4];
    const float* kg_proj_w   = (const float*)d_in[5];
    const float* kg_proj_b   = (const float*)d_in[6];
    const float* w_ih        = (const float*)d_in[7];
    const float* w_hh        = (const float*)d_in[8];
    const float* b_ih        = (const float*)d_in[9];
    const float* b_hh        = (const float*)d_in[10];
    const float* attn_in_w   = (const float*)d_in[11];
    const float* attn_in_b   = (const float*)d_in[12];
    const float* attn_out_w  = (const float*)d_in[13];
    const float* attn_out_b  = (const float*)d_in[14];
    const float* path_proj_w = (const float*)d_in[15];
    const float* path_proj_b = (const float*)d_in[16];
    float* out = (float*)d_out;

    char* ws = (char*)d_ws;
    size_t off = 0;
    auto alloc = [&](size_t bytes) -> void* {
        void* p = ws + off;
        off = (off + bytes + 255) & ~(size_t)255;
        return p;
    };
    _Float16* relT = (_Float16*)alloc((size_t)200 * 256 * 2);
    _Float16* entT = (_Float16*)alloc((size_t)10000 * 256 * 2);
    _Float16* Wp   = (_Float16*)alloc((size_t)256 * 512 * 2);
    _Float16* Wl   = (_Float16*)alloc((size_t)1536 * 2048 * 2);
    float*    gb   = (float*)   alloc((size_t)2048 * 4);
    _Float16* Wq   = (_Float16*)alloc((size_t)512 * 1536 * 2);
    _Float16* Wo   = (_Float16*)alloc((size_t)512 * 512 * 2);
    _Float16* Wpp  = (_Float16*)alloc((size_t)512 * 512 * 2);
    _Float16* xh   = (_Float16*)alloc((size_t)NPK * 1536 * 2);  // [x_t | h]
    _Float16* g    = (_Float16*)alloc((size_t)NPK * 2048 * 2);  // gates
    float*    cst  = (float*)   alloc((size_t)NPK * 512 * 4);   // cell state
    _Float16* pemb = (_Float16*)alloc((size_t)NPK * 512 * 2);   // path_emb
    _Float16* qkv  = (_Float16*)alloc((size_t)NPK * 1536 * 2);
    _Float16* ctx  = (_Float16*)alloc((size_t)NPK * 512 * 2);
    _Float16* aob  = (_Float16*)alloc((size_t)NPK * 512 * 2);
    _Float16* agg  = (_Float16*)alloc((size_t)P_PAIRS * 512 * 2);

    const int T = 256;
    // ---- weight / table prep ----
    cvt_f32_f16<<<(200 * 256 + T - 1) / T, T, 0, stream>>>(rel_table, relT, 200 * 256);
    cvt_f32_f16<<<(10000 * 256 + T - 1) / T, T, 0, stream>>>(ent_table, entT, 10000 * 256);
    tcvt_f32_f16<<<(512 * 256 + T - 1) / T, T, 0, stream>>>(kg_proj_w, Wp, 512, 256);
    build_lstm_B<<<(1536 * 2048 + T - 1) / T, T, 0, stream>>>(w_ih, w_hh, Wl);
    vadd_f32<<<(2048 + T - 1) / T, T, 0, stream>>>(b_ih, b_hh, gb, 2048);
    tcvt_f32_f16<<<(1536 * 512 + T - 1) / T, T, 0, stream>>>(attn_in_w, Wq, 1536, 512);
    tcvt_f32_f16<<<(512 * 512 + T - 1) / T, T, 0, stream>>>(attn_out_w, Wo, 512, 512);
    tcvt_f32_f16<<<(512 * 512 + T - 1) / T, T, 0, stream>>>(path_proj_w, Wpp, 512, 512);
    init_state<<<(NPK * H_DIM + T - 1) / T, T, 0, stream>>>(xh, cst, pemb);

    dim3 blk(256);
    const int mb = NPK / TILE_M;   // 160
    // ---- LSTM over L steps: project step-t embeddings, fused-gate GEMM ----
    for (int t = 0; t < L_STEPS; ++t) {
        // rel_p -> xh[:, 0:512]
        wmma_gemm_f16<<<dim3(512 / TILE_N, mb), blk, 0, stream>>>(
            relT, 256, rel_idx, L_STEPS, t, Wp, 512,
            kg_proj_b, nullptr, xh, nullptr, 1536, 0, 256);
        // ent_p -> xh[:, 512:1024]
        wmma_gemm_f16<<<dim3(512 / TILE_N, mb), blk, 0, stream>>>(
            entT, 256, ent_idx, L_STEPS, t, Wp, 512,
            kg_proj_b, nullptr, xh, nullptr, 1536, 512, 256);
        // gates = [x_t | h] @ [w_ih;w_hh]^T + (b_ih+b_hh)
        wmma_gemm_f16<<<dim3(2048 / TILE_N, mb), blk, 0, stream>>>(
            xh, 1536, nullptr, 0, 0, Wl, 2048,
            gb, nullptr, g, nullptr, 2048, 0, 1536);
        lstm_cell<<<(NPK * H_DIM + T - 1) / T, T, 0, stream>>>(
            g, cst, xh, pemb, path_lens, t);
    }

    // ---- attention ----
    wmma_gemm_f16<<<dim3(1536 / TILE_N, mb), blk, 0, stream>>>(
        pemb, 512, nullptr, 0, 0, Wq, 1536,
        attn_in_b, nullptr, qkv, nullptr, 1536, 0, 512);
    attn_kernel<<<dim3(P_PAIRS, 4), 128, 0, stream>>>(qkv, ctx);
    wmma_gemm_f16<<<dim3(512 / TILE_N, mb), blk, 0, stream>>>(
        ctx, 512, nullptr, 0, 0, Wo, 512,
        attn_out_b, nullptr, aob, nullptr, 512, 0, 512);

    // ---- mean over K + final projection (f32 out) ----
    mean_k<<<(P_PAIRS * H_DIM + T - 1) / T, T, 0, stream>>>(aob, agg);
    wmma_gemm_f16<<<dim3(512 / TILE_N, P_PAIRS / TILE_M), blk, 0, stream>>>(
        agg, 512, nullptr, 0, 0, Wpp, 512,
        path_proj_b, nullptr, nullptr, out, 512, 0, 512);
}
